// StandardRNN_13812614824501
// MI455X (gfx1250) — matrix operand order repaired and checked by reading, not compiled, not run on hardware
//
#include <hip/hip_runtime.h>
#include <math.h>

#define T_ 512
#define B_ 64
#define IN_ 256
#define H_ 512
#define OUT_ 128

typedef __attribute__((ext_vector_type(16))) __bf16 v16bf;
typedef __attribute__((ext_vector_type(8)))  float  v8f;
typedef __attribute__((ext_vector_type(4)))  float  f4;
typedef __attribute__((ext_vector_type(4)))  unsigned int uint4v;

union BF16x16 { v16bf v; uint4v q[2]; unsigned short s[16]; };
union F32x8   { v8f v; f4 h[2]; };

__device__ __forceinline__ unsigned short f2bf(float f) {
  unsigned u = __float_as_uint(f);
  u += 0x7fffu + ((u >> 16) & 1u);   // round-to-nearest-even
  return (unsigned short)(u >> 16);
}

__device__ __forceinline__ v8f wmma_bf16(v16bf a, v16bf b, v8f c) {
  return __builtin_amdgcn_wmma_f32_16x16x32_bf16(false, a, false, b, (short)0, c, false, false);
}

// ---- CDNA5 async global->LDS copy (ASYNCcnt path), with safe fallback ----
// Builtin signature (from clang diagnostic): (int4vec AS1*, int4vec AS3*, Ii offset, Ii cpol)
__device__ __forceinline__ void async_cp16(const uint4v* g, uint4v* l) {
#if __has_builtin(__builtin_amdgcn_global_load_async_to_lds_b128)
  typedef int v4i_vs __attribute__((vector_size(16)));
  typedef __attribute__((address_space(1))) v4i_vs* gp_t;
  typedef __attribute__((address_space(3))) v4i_vs* lp_t;
  __builtin_amdgcn_global_load_async_to_lds_b128((gp_t)g, (lp_t)l, 0, 0);
#else
  *l = *g;
#endif
}

__device__ __forceinline__ void wait_async0() {
#if __has_builtin(__builtin_amdgcn_global_load_async_to_lds_b128)
#  if __has_builtin(__builtin_amdgcn_s_wait_asynccnt)
  __builtin_amdgcn_s_wait_asynccnt(0);
#  else
  asm volatile("s_wait_asynccnt 0" ::: "memory");
#  endif
#endif
}

// ---------------- conversion kernels ----------------

__global__ void k_f2bf(const float* __restrict__ in, unsigned short* __restrict__ out, int n) {
  int i = blockIdx.x * blockDim.x + threadIdx.x;
  if (i < n) out[i] = f2bf(in[i]);
}

// x[B,T,IN] (f32) -> xbf[T,B,IN] (bf16)
__global__ void k_x_to_bf(const float* __restrict__ x, unsigned short* __restrict__ o) {
  size_t idx = (size_t)blockIdx.x * blockDim.x + threadIdx.x;
  int t = (int)(idx / (B_ * IN_));
  int rem = (int)(idx % (B_ * IN_));
  int b = rem / IN_;
  int i = rem % IN_;
  o[idx] = f2bf(x[(size_t)b * T_ * IN_ + (size_t)t * IN_ + i]);
}

__global__ void k_bias_sum(const float* __restrict__ a, const float* __restrict__ b,
                           float* __restrict__ o, int n) {
  int i = blockIdx.x * blockDim.x + threadIdx.x;
  if (i < n) o[i] = a[i] + b[i];
}

// ---------------- input-projection GEMM ----------------
// out is stored [T, H, B] (f32) so the recurrent kernel's C-fragment loads are contiguous.
// A rows ordered (t,b) row-major [T*B, K]; W row-major [H_, K].
__global__ void k_gemm_xin(const unsigned short* __restrict__ A,
                           const unsigned short* __restrict__ W,
                           const float* __restrict__ bsum,
                           float* __restrict__ out, int K) {
  const int wave = threadIdx.x >> 5;
  const int lane = threadIdx.x & 31;
  const int tile = blockIdx.x * (blockDim.x >> 5) + wave;
  const int nt = tile % (H_ / 16);
  const int mt = tile / (H_ / 16);
  const int nn = lane & 15;          // N index (and A-row index m)
  const int kh = lane >> 4;          // lane half

  v8f acc;
  const float bv = bsum[nt * 16 + nn];
#pragma unroll
  for (int v = 0; v < 8; ++v) acc[v] = bv;

  const unsigned short* arow = A + (size_t)(mt * 16 + nn) * K;   // A row m = lane&15
  const unsigned short* wrow = W + (size_t)(nt * 16 + nn) * K;   // B col n = lane&15

  for (int k0 = 0; k0 < K; k0 += 32) {
    BF16x16 a, b;
    // A (contiguous k order chosen here, same as B): lane half holds contiguous K = 16h..16h+15
    a.q[0] = *(const uint4v*)(arow + k0 + 16 * kh);
    a.q[1] = *(const uint4v*)(arow + k0 + 16 * kh + 8);
    // B (contiguous k order chosen here, same as A): lane half holds contiguous K = 16h..16h+15
    b.q[0] = *(const uint4v*)(wrow + k0 + 16 * kh);
    b.q[1] = *(const uint4v*)(wrow + k0 + 16 * kh + 8);
    acc = wmma_bf16(a.v, b.v, acc);
  }

  // tile rows are mt*16..mt*16+15 of the (t,b) row index; B_=64 so one tile = one t
  const int t  = (mt * 16) / B_;
  const int bb = (mt * 16) % B_;
  float* obase = out + ((size_t)t * H_ + nt * 16 + nn) * B_ + bb + kh * 8;
  F32x8 u; u.v = acc;
  *(f4*)(obase)     = u.h[0];   // M = kh*8 + 0..3  (contiguous along B in [T,H,B])
  *(f4*)(obase + 4) = u.h[1];   // M = kh*8 + 4..7
}

// ---------------- persistent recurrent kernel ----------------
// 16 workgroups x 256 threads. WG g: row-tile mt=g&3, col-tiles (g>>2)*8 .. +7 (one per wave).
// W_hh tile held in VGPRs for all T steps; h_prev rows staged in LDS each step (async copy).

__device__ __forceinline__ void grid_barrier(unsigned int* bar, unsigned int target) {
  __syncthreads();
  if (threadIdx.x == 0) {
    __threadfence();
    atomicAdd(bar, 1u);
    while (__hip_atomic_load(bar, __ATOMIC_ACQUIRE, __HIP_MEMORY_SCOPE_AGENT) < target) {
      __builtin_amdgcn_s_sleep(2);
    }
  }
  __syncthreads();
}

__global__ void k_recurrent(const float* __restrict__ xin,          // [T,H,B] f32
                            const unsigned short* __restrict__ whh, // [H,H] bf16
                            unsigned short* __restrict__ hbuf,      // [2,B,H] bf16 (zeroed)
                            unsigned short* __restrict__ hseq,      // [T,B,H] bf16 or null
                            unsigned short* __restrict__ hlast,     // [B,H] bf16 or null
                            unsigned int* __restrict__ bar) {
  __shared__ __attribute__((aligned(16))) unsigned short hA[16 * H_];  // 16 KB: 16 rows of h_prev

  const int tid  = threadIdx.x;
  const int wave = tid >> 5;
  const int lane = tid & 31;
  const int g    = blockIdx.x;          // 0..15
  const int mt   = g & 3;
  const int nt   = (g >> 2) * 8 + wave;
  const int b0   = mt * 16;
  const int n0   = nt * 16;
  const int nn   = lane & 15;
  const int kh   = lane >> 4;

  // Preload this wave's W_hh tile (16 cols x 512 K) into registers: 16 x v16bf.
  BF16x16 breg[16];
#pragma unroll
  for (int kk = 0; kk < 16; ++kk) {
    const unsigned short* wp = whh + (size_t)(n0 + nn) * H_ + kk * 32 + 16 * kh;
    breg[kk].q[0] = *(const uint4v*)(wp);
    breg[kk].q[1] = *(const uint4v*)(wp + 8);
  }

  unsigned short* hb0 = hbuf;
  unsigned short* hb1 = hbuf + B_ * H_;
  unsigned int phase = 0;

  for (int t = 0; t < T_; ++t) {
    const unsigned short* hprev = (t & 1) ? hb1 : hb0;
    unsigned short*       hcur  = (t & 1) ? hb0 : hb1;

    if (t > 0) { ++phase; grid_barrier(bar, 16u * phase); }

    // stage h_prev rows b0..b0+15 into LDS (16 KB) via async global->LDS copies
    {
      const uint4v* src = (const uint4v*)(hprev + (size_t)b0 * H_);
      uint4v* dst = (uint4v*)hA;
      for (int i = tid; i < (16 * H_) / 8; i += 256) async_cp16(src + i, dst + i);
      wait_async0();
    }
    // prefetch next step's xin block for this lane (hidden behind WMMAs)
    if (t + 1 < T_)
      __builtin_prefetch(xin + ((size_t)(t + 1) * H_ + n0 + nn) * B_ + b0, 0, 1);
    __syncthreads();

    // C init from xin[T,H,B]: 8 consecutive floats along B per lane
    F32x8 cu;
    const float* cbase = xin + ((size_t)t * H_ + n0 + nn) * B_ + b0 + kh * 8;
    cu.h[0] = *(const f4*)(cbase);
    cu.h[1] = *(const f4*)(cbase + 4);
    v8f acc = cu.v;

    // K loop: A from LDS, B from registers
#pragma unroll
    for (int kk = 0; kk < 16; ++kk) {
      BF16x16 a;
      const unsigned short* ap = hA + (size_t)nn * H_ + kk * 32 + 16 * kh;
      a.q[0] = *(const uint4v*)(ap);
      a.q[1] = *(const uint4v*)(ap + 8);
      acc = wmma_bf16(a.v, breg[kk].v, acc);
    }

    // tanh + store new h
#pragma unroll
    for (int v = 0; v < 8; ++v) {
      const float hv = tanhf(acc[v]);
      const unsigned short s = f2bf(hv);
      const size_t roff = (size_t)(b0 + kh * 8 + v) * H_ + n0 + nn;
      hcur[roff] = s;
      if (hseq)  hseq[(size_t)t * B_ * H_ + roff] = s;
      if (hlast && t == T_ - 1) hlast[roff] = s;
    }
  }
}

// ---------------- FC: out[64,128] = h[64,512] . fcw[128,512]^T + fcb ----------------
__global__ void k_fc(const unsigned short* __restrict__ h,
                     const unsigned short* __restrict__ fcw,
                     const float* __restrict__ fcb,
                     float* __restrict__ out) {
  const int wave = threadIdx.x >> 5;
  const int lane = threadIdx.x & 31;
  const int tile = blockIdx.x * 8 + wave;   // 32 tiles: 4 M x 8 N
  const int nt = tile & 7;
  const int mt = tile >> 3;
  const int nn = lane & 15;
  const int kh = lane >> 4;

  v8f acc;
  const float bv = fcb[nt * 16 + nn];
#pragma unroll
  for (int v = 0; v < 8; ++v) acc[v] = bv;

  const unsigned short* arow = h   + (size_t)(mt * 16 + nn) * H_;
  const unsigned short* wrow = fcw + (size_t)(nt * 16 + nn) * H_;
#pragma unroll
  for (int kk = 0; kk < 16; ++kk) {
    BF16x16 a, b;
    a.q[0] = *(const uint4v*)(arow + kk * 32 + 16 * kh);
    a.q[1] = *(const uint4v*)(arow + kk * 32 + 16 * kh + 8);
    b.q[0] = *(const uint4v*)(wrow + kk * 32 + 16 * kh);
    b.q[1] = *(const uint4v*)(wrow + kk * 32 + 16 * kh + 8);
    acc = wmma_bf16(a.v, b.v, acc);
  }
#pragma unroll
  for (int v = 0; v < 8; ++v)
    out[(size_t)(mt * 16 + kh * 8 + v) * OUT_ + nt * 16 + nn] = acc[v];
}

// ---------------- host launch ----------------

extern "C" void kernel_launch(void* const* d_in, const int* in_sizes, int n_in,
                              void* d_out, int out_size, void* d_ws, size_t ws_size,
                              hipStream_t stream) {
  (void)in_sizes; (void)n_in; (void)out_size; (void)ws_size;

  const float* x     = (const float*)d_in[0];
  const float* w_ih0 = (const float*)d_in[1];
  const float* w_hh0 = (const float*)d_in[2];
  const float* b_ih0 = (const float*)d_in[3];
  const float* b_hh0 = (const float*)d_in[4];
  const float* w_ih1 = (const float*)d_in[5];
  const float* w_hh1 = (const float*)d_in[6];
  const float* b_ih1 = (const float*)d_in[7];
  const float* b_hh1 = (const float*)d_in[8];
  const float* fc_w  = (const float*)d_in[9];
  const float* fc_b  = (const float*)d_in[10];

  char* ws = (char*)d_ws;
  size_t off = 0;
  auto alloc = [&](size_t bytes) -> void* {
    void* p = ws + off;
    off = (off + bytes + 255) & ~(size_t)255;
    return p;
  };

  unsigned short* xbf    = (unsigned short*)alloc((size_t)T_ * B_ * IN_ * 2);
  unsigned short* wih0bf = (unsigned short*)alloc((size_t)H_ * IN_ * 2);
  unsigned short* whh0bf = (unsigned short*)alloc((size_t)H_ * H_ * 2);
  unsigned short* wih1bf = (unsigned short*)alloc((size_t)H_ * H_ * 2);
  unsigned short* whh1bf = (unsigned short*)alloc((size_t)H_ * H_ * 2);
  unsigned short* fcwbf  = (unsigned short*)alloc((size_t)OUT_ * H_ * 2);
  float*          bsum0  = (float*)alloc((size_t)H_ * 4);
  float*          bsum1  = (float*)alloc((size_t)H_ * 4);
  float*          xin    = (float*)alloc((size_t)T_ * B_ * H_ * 4);   // [T,H,B]
  unsigned short* h1seq  = (unsigned short*)alloc((size_t)T_ * B_ * H_ * 2);
  unsigned short* hbuf   = (unsigned short*)alloc((size_t)2 * B_ * H_ * 2);
  unsigned short* h2last = (unsigned short*)alloc((size_t)B_ * H_ * 2);
  unsigned int*   bar    = (unsigned int*)alloc(256);

  const int thr = 256;

  // convert weights + input
  k_f2bf<<<(H_ * IN_) / thr, thr, 0, stream>>>(w_ih0, wih0bf, H_ * IN_);
  k_f2bf<<<(H_ * H_) / thr, thr, 0, stream>>>(w_hh0, whh0bf, H_ * H_);
  k_f2bf<<<(H_ * H_) / thr, thr, 0, stream>>>(w_ih1, wih1bf, H_ * H_);
  k_f2bf<<<(H_ * H_) / thr, thr, 0, stream>>>(w_hh1, whh1bf, H_ * H_);
  k_f2bf<<<(OUT_ * H_) / thr, thr, 0, stream>>>(fc_w, fcwbf, OUT_ * H_);
  k_x_to_bf<<<(T_ * B_ * IN_) / thr, thr, 0, stream>>>(x, xbf);
  k_bias_sum<<<(H_ + thr - 1) / thr, thr, 0, stream>>>(b_ih0, b_hh0, bsum0, H_);
  k_bias_sum<<<(H_ + thr - 1) / thr, thr, 0, stream>>>(b_ih1, b_hh1, bsum1, H_);

  const int tiles = (T_ * B_ / 16) * (H_ / 16);  // 65536 tiles, 8 waves/block

  // layer 0
  k_gemm_xin<<<tiles / 8, thr, 0, stream>>>(xbf, wih0bf, bsum0, xin, IN_);
  (void)hipMemsetAsync(hbuf, 0, (size_t)2 * B_ * H_ * 2, stream);
  (void)hipMemsetAsync(bar, 0, 256, stream);
  k_recurrent<<<16, thr, 0, stream>>>(xin, whh0bf, hbuf, h1seq, nullptr, bar);

  // layer 1
  k_gemm_xin<<<tiles / 8, thr, 0, stream>>>(h1seq, wih1bf, bsum1, xin, H_);
  (void)hipMemsetAsync(hbuf, 0, (size_t)2 * B_ * H_ * 2, stream);
  (void)hipMemsetAsync(bar, 0, 256, stream);
  k_recurrent<<<16, thr, 0, stream>>>(xin, whh1bf, hbuf, nullptr, h2last, bar);

  // final FC
  k_fc<<<4, thr, 0, stream>>>(h2last, fcwbf, fc_b, (float*)d_out);
}
